// NMNIST_86088324481701
// MI455X (gfx1250) — compile-verified, matrix operand order checked by reading, not run on hardware
//
#include <hip/hip_runtime.h>
#include <math.h>

// ---- problem constants (from reference) ----
#define TSTEPS   300
#define BATCH    256
#define TB       (TSTEPS * BATCH)   // 76800
#define INFEAT   578                // 2*17*17
#define KPAD     608                // 19 * 32
#define KCH      19                 // k-chunks of 32
#define H1       20
#define H2       10
#define IMG_STRIDE 2312             // 2*34*34 floats per (t,b)
#define CH_STRIDE  1156             // 34*34
#define ROWS     128                // (t,b) rows per block

typedef _Float16 v16h __attribute__((ext_vector_type(16)));
typedef _Float16 v8h  __attribute__((ext_vector_type(8)));
typedef float    v8f  __attribute__((ext_vector_type(8)));
typedef float    v4f  __attribute__((ext_vector_type(4)));

// =====================================================================
// Kernel 1: fused stride-2 gather + f16 conversion + WMMA GEMM
//   cur1[t*B+b][n] = sum_k xs[t,b,k] * W1[n][k] + b1[n]
// Block: 256 threads = 8 waves; each wave computes a 16x32 (M x Npad) tile.
// =====================================================================
__global__ __launch_bounds__(256) void snn_gemm1(
    const float* __restrict__ x,      // [T,B,2,34,34]
    const float* __restrict__ W1,     // [20,578]
    const float* __restrict__ b1,     // [20]
    float* __restrict__ cur1)         // [TB,20]
{
  __shared__ __align__(16) _Float16 sX[ROWS][KPAD];  // 155,648 B
  __shared__ __align__(16) _Float16 sW[32][KPAD];    //  38,912 B

  const int tid = threadIdx.x;
  const long rowStart = (long)blockIdx.x * ROWS;

  // ---- phase A: zero W pad + X K-tails ----
  {
    v8h z = {};
    for (int i = tid; i < 32 * KPAD / 8; i += 256)
      ((v8h*)sW)[i] = z;
    // zero sX[:, 576..607] (cols 576/577 get refilled in phase B)
    for (int i = tid; i < ROWS * 16; i += 256) {
      int row = i >> 4;
      int k   = 576 + ((i & 15) << 1);
      *(unsigned int*)&sX[row][k] = 0u;
    }
  }
  __syncthreads();

  // ---- phase B: gather x[:, :, :, ::2, ::2] -> f16 LDS, and W1 -> f16 LDS ----
  // One task per (row, channel, out-row): 128*2*17 = 4352 tasks
  for (int tau = tid; tau < ROWS * 34; tau += 256) {
    int row = tau / 34;
    int s   = tau - row * 34;
    int c   = s / 17;
    int r   = s - c * 17;
    const float* src = x + (rowStart + row) * (long)IMG_STRIDE
                         + c * CH_STRIDE + (2 * r) * 34;
    _Float16* dst = &sX[row][c * 289 + r * 17];
    #pragma unroll
    for (int q = 0; q < 8; ++q) {           // cols 0..31 as float4s
      v4f v = ((const v4f*)src)[q];
      dst[2 * q]     = (_Float16)v.x;       // col 4q   -> cc 2q
      dst[2 * q + 1] = (_Float16)v.z;       // col 4q+2 -> cc 2q+1
    }
    dst[16] = (_Float16)src[32];            // col 32 -> cc 16
  }
  for (int i = tid; i < H1 * INFEAT; i += 256) {
    int n = i / INFEAT;
    int k = i - n * INFEAT;
    sW[n][k] = (_Float16)W1[i];
  }
  __syncthreads();

  // ---- phase C: WMMA compute ----
  const int lane = tid & 31;
  const int wave = tid >> 5;
  const int h    = lane >> 4;    // lane half: selects K sub-block
  const int l16  = lane & 15;
  const int arow = wave * 16 + l16;

  v8f acc0 = {}, acc1 = {};
  for (int kk = 0; kk < KCH; ++kk) {
    const int kb = kk * 32 + h * 8;
    v16h a, b0, b1v;
    // A fragment: row = arow, K = kb..kb+7 and kb+16..kb+23 (ISA 16-bit A layout)
    *((v8h*)&a)       = *(const v8h*)&sX[arow][kb];
    *((v8h*)&a + 1)   = *(const v8h*)&sX[arow][kb + 16];
    // B fragments: column n = l16 (tile0) / 16+l16 (tile1), same K striping
    *((v8h*)&b0)      = *(const v8h*)&sW[l16][kb];
    *((v8h*)&b0 + 1)  = *(const v8h*)&sW[l16][kb + 16];
    *((v8h*)&b1v)     = *(const v8h*)&sW[16 + l16][kb];
    *((v8h*)&b1v + 1) = *(const v8h*)&sW[16 + l16][kb + 16];

    acc0 = __builtin_amdgcn_wmma_f32_16x16x32_f16(false, a, false, b0,
                                                  (short)0, acc0, false, false);
    acc1 = __builtin_amdgcn_wmma_f32_16x16x32_f16(false, a, false, b1v,
                                                  (short)0, acc1, false, false);
  }

  // ---- epilogue: D layout -> cur1[row][n], add bias ----
  // D VGPR i: lanes 0-15 hold M=i, lanes 16-31 hold M=i+8; N = l16
  const float bias0 = b1[l16];
  #pragma unroll
  for (int i = 0; i < 8; ++i) {
    long gr = rowStart + wave * 16 + h * 8 + i;
    cur1[gr * H1 + l16] = acc0[i] + bias0;
  }
  if (l16 < 4) {
    const float bias1 = b1[16 + l16];
    #pragma unroll
    for (int i = 0; i < 8; ++i) {
      long gr = rowStart + wave * 16 + h * 8 + i;
      cur1[gr * H1 + 16 + l16] = acc1[i] + bias1;
    }
  }
}

// =====================================================================
// Kernel 2: sequential LIF scan + layer-2 linear + spike-count softmax.
// One thread per batch element; 8 blocks x 32 lanes.
// =====================================================================
__global__ __launch_bounds__(32) void snn_scan(
    const float* __restrict__ cur1,   // [TB,20]
    const float* __restrict__ W2,     // [10,20]
    const float* __restrict__ b2,     // [10]
    const float* __restrict__ beta1,  // [20]
    const float* __restrict__ th1,    // [20]
    const float* __restrict__ beta2,  // [10]
    const float* __restrict__ th2,    // [10]
    float* __restrict__ out)          // [B,10]
{
  __shared__ float w2s[H2 * H1];
  __shared__ float b2s[H2], bc2s[H2], th2s[H2];
  __shared__ float bc1s[H1], th1s[H1];

  const int tid = threadIdx.x;
  for (int i = tid; i < H2 * H1; i += 32) w2s[i] = W2[i];
  if (tid < H2) {
    b2s[tid]  = b2[tid];
    float bb  = beta2[tid];
    bc2s[tid] = bb < 0.f ? 0.f : (bb > 1.f ? 1.f : bb);
    th2s[tid] = th2[tid];
  }
  if (tid < H1) {
    float bb  = beta1[tid];
    bc1s[tid] = bb < 0.f ? 0.f : (bb > 1.f ? 1.f : bb);
    th1s[tid] = th1[tid];
  }
  __syncthreads();

  const int b = blockIdx.x * 32 + tid;

  float mem1[H1], mem2[H2], sum2[H2];
  #pragma unroll
  for (int i = 0; i < H1; ++i) mem1[i] = 0.f;
  #pragma unroll
  for (int i = 0; i < H2; ++i) { mem2[i] = 0.f; sum2[i] = 0.f; }

  for (int t = 0; t < TSTEPS; ++t) {
    const v4f* p = (const v4f*)(cur1 + ((size_t)t * BATCH + b) * H1);
    float c1[H1];
    #pragma unroll
    for (int q = 0; q < 5; ++q) {
      v4f v = p[q];
      c1[4 * q] = v.x; c1[4 * q + 1] = v.y; c1[4 * q + 2] = v.z; c1[4 * q + 3] = v.w;
    }

    float spk1[H1];
    #pragma unroll
    for (int hh = 0; hh < H1; ++hh) {
      float th = th1s[hh];
      float r  = (mem1[hh] > th) ? th : 0.f;       // detached reset (prev mem)
      mem1[hh] = fmaf(bc1s[hh], mem1[hh], c1[hh] - r);
      spk1[hh] = (mem1[hh] > th) ? 1.f : 0.f;      // Heaviside(mem - th)
    }

    #pragma unroll
    for (int j = 0; j < H2; ++j) {
      float c2 = b2s[j];
      #pragma unroll
      for (int hh = 0; hh < H1; ++hh)
        c2 = fmaf(spk1[hh], w2s[j * H1 + hh], c2);
      float th = th2s[j];
      float r  = (mem2[j] > th) ? th : 0.f;
      mem2[j]  = fmaf(bc2s[j], mem2[j], c2 - r);
      sum2[j] += (mem2[j] > th) ? 1.f : 0.f;
    }
  }

  // softmax over sum2
  float m = sum2[0];
  #pragma unroll
  for (int j = 1; j < H2; ++j) m = sum2[j] > m ? sum2[j] : m;
  float e[H2], s = 0.f;
  #pragma unroll
  for (int j = 0; j < H2; ++j) { e[j] = __expf(sum2[j] - m); s += e[j]; }
  float inv = 1.f / s;
  #pragma unroll
  for (int j = 0; j < H2; ++j) out[(size_t)b * H2 + j] = e[j] * inv;
}

// =====================================================================
extern "C" void kernel_launch(void* const* d_in, const int* in_sizes, int n_in,
                              void* d_out, int out_size, void* d_ws, size_t ws_size,
                              hipStream_t stream) {
  const float* x     = (const float*)d_in[0];
  const float* W1    = (const float*)d_in[1];
  const float* b1    = (const float*)d_in[2];
  const float* W2    = (const float*)d_in[3];
  const float* b2    = (const float*)d_in[4];
  const float* beta1 = (const float*)d_in[5];
  const float* th1   = (const float*)d_in[6];
  const float* beta2 = (const float*)d_in[7];
  const float* th2   = (const float*)d_in[8];
  float* out  = (float*)d_out;
  float* cur1 = (float*)d_ws;                 // TB*20 floats = 6.14 MB

  snn_gemm1<<<TB / ROWS, 256, 0, stream>>>(x, W1, b1, cur1);
  snn_scan<<<BATCH / 32, 32, 0, stream>>>(cur1, W2, b2, beta1, th1, beta2, th2, out);
}
